// PointTransformer_850403525317
// MI455X (gfx1250) — matrix operand order repaired and checked by reading, not compiled, not run on hardware
//
#include <hip/hip_runtime.h>
#include <hip/hip_bf16.h>
#include <math.h>

typedef __attribute__((ext_vector_type(16))) _Float16 f16x16;
typedef __attribute__((ext_vector_type(8)))  _Float16 f16x8;
typedef __attribute__((ext_vector_type(8)))  float    fx8;

#define NPTS 512
#define DMODEL 256
#define NHEAD 8
#define HDIM 32
#define HIDP 688   /* 682 padded to multiple of 16 */
#define HIDR 682

// ---------------------------------------------------------------------------
// Generic f16-in / f32-acc WMMA GEMM:  C = alpha * (A @ W^T) + beta*C + bias
//   A: [M][K] f16 row-major with row stride lda (elements)
//   W: [N][K] f16 row-major with row stride ldb (elements)  -> B_wmma[k][n]=W[n][k]
//   C: [M][N] f32 with row stride ldc
// One wave (32 lanes) produces one 16x16 tile. K must be a multiple of 32,
// M,N multiples of 16, lda/ldb multiples of 8 (16B-aligned vector loads).
// VGPR layouts follow CDNA5 ISA 7.12.2 (wave32).
// ---------------------------------------------------------------------------
__global__ void gemm_wmma_f16(const _Float16* __restrict__ A, int lda,
                              const _Float16* __restrict__ W, int ldb,
                              float* __restrict__ C, int ldc,
                              int M, int N, int K,
                              float alpha, float beta,
                              const float* __restrict__ bias, int bias_n)
{
    const int wave = threadIdx.x >> 5;
    const int lane = threadIdx.x & 31;
    const int tiles_n = N >> 4;
    const int tile = blockIdx.x * (blockDim.x >> 5) + wave;
    if (tile >= (M >> 4) * tiles_n) return;
    const int tm = tile / tiles_n;
    const int tn = tile % tiles_n;

    const int l15 = lane & 15;      // A: row-in-tile, B: col-in-tile
    const int g   = lane >> 4;      // half-wave group

    const _Float16* aRow = A + (size_t)((tm << 4) + l15) * lda;
    const _Float16* bRow = W + (size_t)((tn << 4) + l15) * ldb;

    fx8 acc = {};
    for (int kk = 0; kk < K; kk += 32) {
        // A 16x32 f16: lane l15 (+16g) supplies K in [g*8,g*8+8) and [16+g*8,16+g*8+8)
        f16x8 a0 = *reinterpret_cast<const f16x8*>(aRow + kk + g * 8);
        f16x8 a1 = *reinterpret_cast<const f16x8*>(aRow + kk + 16 + g * 8);
        // B 32x16 f16: lane supplies column n, K in [g*8,g*8+8) and [16+g*8,16+g*8+8)
        f16x8 b0 = *reinterpret_cast<const f16x8*>(bRow + kk + g * 8);
        f16x8 b1 = *reinterpret_cast<const f16x8*>(bRow + kk + 16 + g * 8);
        if (kk + 32 < K) {
            __builtin_prefetch(aRow + kk + 32, 0, 1);
            __builtin_prefetch(bRow + kk + 32, 0, 1);
        }
        f16x16 av = __builtin_shufflevector(a0, a1, 0,1,2,3,4,5,6,7,8,9,10,11,12,13,14,15);
        f16x16 bv = __builtin_shufflevector(b0, b1, 0,1,2,3,4,5,6,7,8,9,10,11,12,13,14,15);
        acc = __builtin_amdgcn_wmma_f32_16x16x32_f16(false, av, false, bv,
                                                     (short)0, acc, false, false);
    }

    const int n = (tn << 4) + l15;
    float bv = 0.f;
    if (bias) bv = (n < bias_n) ? bias[n] : 0.f;
#pragma unroll
    for (int r = 0; r < 8; ++r) {
        const int m = (tm << 4) + (g << 3) + r;     // C/D layout: row = r + 8*g
        const size_t idx = (size_t)m * ldc + n;
        float v = alpha * acc[r] + bv;
        if (beta != 0.f) v += beta * C[idx];
        C[idx] = v;
    }
}

// ---------------------------------------------------------------------------
// LayerNorm over D=256, one block per row, writes f16 for WMMA consumption.
// ---------------------------------------------------------------------------
__global__ void layernorm_kernel(const float* __restrict__ x, const float* __restrict__ w,
                                 const float* __restrict__ b, _Float16* __restrict__ yh)
{
    __shared__ float red[DMODEL];
    const int i = blockIdx.x, d = threadIdx.x;
    const float v = x[(size_t)i * DMODEL + d];
    red[d] = v; __syncthreads();
    for (int s = 128; s > 0; s >>= 1) { if (d < s) red[d] += red[d + s]; __syncthreads(); }
    const float mu = red[0] * (1.f / DMODEL); __syncthreads();
    const float c = v - mu;
    red[d] = c * c; __syncthreads();
    for (int s = 128; s > 0; s >>= 1) { if (d < s) red[d] += red[d + s]; __syncthreads(); }
    const float var = red[0] * (1.f / DMODEL);
    yh[(size_t)i * DMODEL + d] = (_Float16)(c * rsqrtf(var + 1e-5f) * w[d] + b[d]);
}

// f32 -> f16 with zero padding (weights; also used unpadded)
__global__ void cvt_pad_kernel(const float* __restrict__ src, _Float16* __restrict__ dst,
                               int srows, int scols, int drows, int dcols)
{
    const int i = blockIdx.x * blockDim.x + threadIdx.x;
    if (i >= drows * dcols) return;
    const int r = i / dcols, c = i % dcols;
    const float v = (r < srows && c < scols) ? src[(size_t)r * scols + c] : 0.f;
    dst[i] = (_Float16)v;
}

// P[j][k] = coords_j . w[k,:]  for both rel-bias and rel-value first layers
__global__ void coordproj_kernel(const float* __restrict__ coords,
                                 const float* __restrict__ rb1w, const float* __restrict__ rv1w,
                                 float* __restrict__ Prb, float* __restrict__ Urv)
{
    const int j = blockIdx.x, k = threadIdx.x;
    const float c0 = coords[j * 3 + 0], c1 = coords[j * 3 + 1], c2 = coords[j * 3 + 2];
    Prb[(size_t)j * DMODEL + k] = rb1w[k * 3] * c0 + rb1w[k * 3 + 1] * c1 + rb1w[k * 3 + 2] * c2;
    Urv[(size_t)j * DMODEL + k] = rv1w[k * 3] * c0 + rv1w[k * 3 + 1] * c1 + rv1w[k * 3 + 2] * c2;
}

// qkv f32 -> f16 copy (for Q/K WMMA) + per-head transposed V (for attn*V WMMA)
__global__ void qkvcvt_kernel(const float* __restrict__ qkvF, _Float16* __restrict__ qkvh,
                              _Float16* __restrict__ vt)
{
    const int idx = blockIdx.x * blockDim.x + threadIdx.x;   // 512*768
    const int i = idx / 768, c = idx % 768;
    const float v = qkvF[idx];
    qkvh[idx] = (_Float16)v;
    if (c >= 512) vt[(size_t)(c - 512) * NPTS + i] = (_Float16)v;   // vt[d][j]
}

// rel_bias[h,i,j] = sum_k silu(Prb_i[k]+b1[k]-Prb_j[k]) * rb2w[h,k] + rb2b[h]
// One block per query i; thread t owns j = t and j+256.
__global__ void relbias_kernel(const float* __restrict__ Prb, const float* __restrict__ rb1b,
                               const float* __restrict__ rb2w, const float* __restrict__ rb2b,
                               float* __restrict__ scores)
{
    __shared__ float sPbi[DMODEL];
    __shared__ float sW[NHEAD * DMODEL];
    const int i = blockIdx.x, t = threadIdx.x;
    sPbi[t] = Prb[(size_t)i * DMODEL + t] + rb1b[t];
#pragma unroll
    for (int h = 0; h < NHEAD; ++h) sW[h * DMODEL + t] = rb2w[h * DMODEL + t];
    __syncthreads();

    for (int jj = 0; jj < 2; ++jj) {
        const int j = t + jj * 256;
        const float* Pj = Prb + (size_t)j * DMODEL;
        float acc[NHEAD] = {0.f, 0.f, 0.f, 0.f, 0.f, 0.f, 0.f, 0.f};
        for (int k = 0; k < DMODEL; ++k) {
            const float z = sPbi[k] - Pj[k];
            const float s = z / (1.f + __expf(-z));
#pragma unroll
            for (int h = 0; h < NHEAD; ++h) acc[h] += s * sW[h * DMODEL + k];
        }
#pragma unroll
        for (int h = 0; h < NHEAD; ++h)
            scores[((size_t)h * NPTS + i) * NPTS + j] = acc[h] + rb2b[h];
    }
}

// softmax over j (row length 512) in place; also write f16 weights for WMMA
__global__ void softmax_kernel(float* __restrict__ buf, _Float16* __restrict__ wh)
{
    __shared__ float red[256];
    const int t = threadIdx.x;
    const size_t base = (size_t)blockIdx.x * NPTS;
    const float v0 = buf[base + t], v1 = buf[base + t + 256];
    red[t] = fmaxf(v0, v1); __syncthreads();
    for (int s = 128; s > 0; s >>= 1) { if (t < s) red[t] = fmaxf(red[t], red[t + s]); __syncthreads(); }
    const float m = red[0]; __syncthreads();
    const float e0 = __expf(v0 - m), e1 = __expf(v1 - m);
    red[t] = e0 + e1; __syncthreads();
    for (int s = 128; s > 0; s >>= 1) { if (t < s) red[t] += red[t + s]; __syncthreads(); }
    const float r = 1.f / red[0];
    buf[base + t] = e0 * r; buf[base + t + 256] = e1 * r;
    wh[base + t] = (_Float16)(e0 * r); wh[base + t + 256] = (_Float16)(e1 * r);
}

// T[i,h,k] = sum_j w[h,i,j] * silu(Urv_i[k]+rv1b[k]-Urv_j[k]); block per i, thread per k
__global__ void tsum_kernel(const float* __restrict__ Urv, const float* __restrict__ rv1b,
                            const float* __restrict__ wgt, float* __restrict__ T)
{
    __shared__ float sw[NHEAD * NPTS];   // 16 KB: weights rows for this i, all heads
    const int i = blockIdx.x, k = threadIdx.x;
    for (int r = k; r < NHEAD * NPTS; r += 256)
        sw[r] = wgt[((size_t)(r >> 9) * NPTS + i) * NPTS + (r & 511)];
    __syncthreads();

    const float ub = Urv[(size_t)i * DMODEL + k] + rv1b[k];
    float acc[NHEAD] = {0.f, 0.f, 0.f, 0.f, 0.f, 0.f, 0.f, 0.f};
    for (int j = 0; j < NPTS; ++j) {
        const float z = ub - Urv[(size_t)j * DMODEL + k];
        const float s = z / (1.f + __expf(-z));
#pragma unroll
        for (int h = 0; h < NHEAD; ++h) acc[h] += sw[h * NPTS + j] * s;
    }
#pragma unroll
    for (int h = 0; h < NHEAD; ++h) T[((size_t)i * NHEAD + h) * DMODEL + k] = acc[h];
}

// ctx[i,d] += sum_k T[i, d/32, k] * rv2w[d,k] + rv2b[d]   (softmax weights sum to 1)
__global__ void ctx2_kernel(const float* __restrict__ T, const float* __restrict__ rv2w,
                            const float* __restrict__ rv2b, float* __restrict__ ctx)
{
    const int i = blockIdx.x, d = threadIdx.x;
    const float* trow = T + ((size_t)i * NHEAD + (d >> 5)) * DMODEL;
    const float* wrow = rv2w + (size_t)d * DMODEL;
    float acc = 0.f;
    for (int k = 0; k < DMODEL; ++k) acc += trow[k] * wrow[k];
    ctx[(size_t)i * DMODEL + d] += acc + rv2b[d];
}

__global__ void add_kernel(const float* __restrict__ a, const float* __restrict__ b,
                           float* __restrict__ o, int n)
{
    const int i = blockIdx.x * blockDim.x + threadIdx.x;
    if (i < n) o[i] = a[i] + b[i];
}

__global__ void swiglu_kernel(const float* __restrict__ h1, const float* __restrict__ h2,
                              _Float16* __restrict__ hp, int n)
{
    const int i = blockIdx.x * blockDim.x + threadIdx.x;
    if (i >= n) return;
    const float a = h1[i];
    hp[i] = (_Float16)((a / (1.f + __expf(-a))) * h2[i]);
}

// ---------------------------------------------------------------------------
static inline size_t alignup(size_t x) { return (x + 255) & ~(size_t)255; }

extern "C" void kernel_launch(void* const* d_in, const int* in_sizes, int n_in,
                              void* d_out, int out_size, void* d_ws, size_t ws_size,
                              hipStream_t stream)
{
    const float* x      = (const float*)d_in[0];
    const float* coords = (const float*)d_in[1];
    const float* ln1_w  = (const float*)d_in[2];
    const float* ln1_b  = (const float*)d_in[3];
    const float* ln2_w  = (const float*)d_in[4];
    const float* ln2_b  = (const float*)d_in[5];
    const float* qkv_w  = (const float*)d_in[6];
    const float* qkv_b  = (const float*)d_in[7];
    const float* rb1_w  = (const float*)d_in[8];
    const float* rb1_b  = (const float*)d_in[9];
    const float* rb2_w  = (const float*)d_in[10];
    const float* rb2_b  = (const float*)d_in[11];
    const float* rv1_w  = (const float*)d_in[12];
    const float* rv1_b  = (const float*)d_in[13];
    const float* rv2_w  = (const float*)d_in[14];
    const float* rv2_b  = (const float*)d_in[15];
    const float* ffn_w1 = (const float*)d_in[16];
    const float* ffn_b1 = (const float*)d_in[17];
    const float* ffn_w2 = (const float*)d_in[18];
    const float* ffn_b2 = (const float*)d_in[19];
    const float* ffn_w3 = (const float*)d_in[20];
    const float* ffn_b3 = (const float*)d_in[21];
    float* out = (float*)d_out;

    // workspace carving (bump allocator, 256B aligned)
    char* p = (char*)d_ws;
    auto carve = [&](size_t bytes) { char* r = p; p += alignup(bytes); return r; };
    _Float16* qkvw_h = (_Float16*)carve(768 * 256 * 2);
    _Float16* w1h    = (_Float16*)carve(HIDP * 256 * 2);
    _Float16* w2h    = (_Float16*)carve(HIDP * 256 * 2);
    _Float16* w3h    = (_Float16*)carve(256 * HIDP * 2);
    _Float16* xnh    = (_Float16*)carve(NPTS * DMODEL * 2);
    _Float16* xn2h   = (_Float16*)carve(NPTS * DMODEL * 2);
    float*    qkvF   = (float*)carve(NPTS * 768 * 4);
    _Float16* qkvh   = (_Float16*)carve(NPTS * 768 * 2);
    _Float16* vt     = (_Float16*)carve(DMODEL * NPTS * 2);
    float*    Prb    = (float*)carve(NPTS * DMODEL * 4);
    float*    Urv    = (float*)carve(NPTS * DMODEL * 4);
    float*    scores = (float*)carve((size_t)NHEAD * NPTS * NPTS * 4);   // rel_bias -> scores -> weights
    _Float16* wh     = (_Float16*)carve((size_t)NHEAD * NPTS * NPTS * 2);
    float*    T      = (float*)carve((size_t)NPTS * NHEAD * DMODEL * 4);
    float*    ctx    = (float*)carve(NPTS * DMODEL * 4);
    float*    x2     = (float*)carve(NPTS * DMODEL * 4);
    float*    h1     = (float*)carve(NPTS * HIDP * 4);
    float*    h2     = (float*)carve(NPTS * HIDP * 4);
    _Float16* hph    = (_Float16*)carve(NPTS * HIDP * 2);
    float*    ffnO   = (float*)carve(NPTS * DMODEL * 4);
    (void)ws_size; (void)n_in; (void)in_sizes; (void)out_size;

    auto gemm = [&](const _Float16* A, int lda, const _Float16* W, int ldb,
                    float* C, int ldc, int M, int N, int K,
                    float alpha, float beta, const float* bias, int bias_n) {
        int tiles = (M / 16) * (N / 16);
        gemm_wmma_f16<<<dim3((tiles + 3) / 4), dim3(128), 0, stream>>>(
            A, lda, W, ldb, C, ldc, M, N, K, alpha, beta, bias, bias_n);
    };

    // --- weight conversions (f32 -> f16, padded where needed) ---
    cvt_pad_kernel<<<dim3((768 * 256 + 255) / 256), dim3(256), 0, stream>>>(qkv_w, qkvw_h, 768, 256, 768, 256);
    cvt_pad_kernel<<<dim3((HIDP * 256 + 255) / 256), dim3(256), 0, stream>>>(ffn_w1, w1h, HIDR, 256, HIDP, 256);
    cvt_pad_kernel<<<dim3((HIDP * 256 + 255) / 256), dim3(256), 0, stream>>>(ffn_w2, w2h, HIDR, 256, HIDP, 256);
    cvt_pad_kernel<<<dim3((256 * HIDP + 255) / 256), dim3(256), 0, stream>>>(ffn_w3, w3h, 256, HIDR, 256, HIDP);

    // --- rank-structured coordinate projections (rel @ W.T == P_i - P_j) ---
    coordproj_kernel<<<dim3(NPTS), dim3(DMODEL), 0, stream>>>(coords, rb1_w, rv1_w, Prb, Urv);

    // --- LN1 + QKV projection (WMMA) ---
    layernorm_kernel<<<dim3(NPTS), dim3(DMODEL), 0, stream>>>(x, ln1_w, ln1_b, xnh);
    gemm(xnh, 256, qkvw_h, 256, qkvF, 768, NPTS, 768, 256, 1.f, 0.f, qkv_b, 768);
    qkvcvt_kernel<<<dim3(NPTS * 768 / 256), dim3(256), 0, stream>>>(qkvF, qkvh, vt);

    // --- rel_bias into score buffer, then scores += QK^T/sqrt(hd) (WMMA, K=32) ---
    relbias_kernel<<<dim3(NPTS), dim3(256), 0, stream>>>(Prb, rb1_b, rb2_w, rb2_b, scores);
    for (int h = 0; h < NHEAD; ++h)
        gemm(qkvh + h * HDIM, 768, qkvh + 512 + h * HDIM, 768,
             scores + (size_t)h * NPTS * NPTS, NPTS,
             NPTS, NPTS, HDIM, 0.17677669529663689f, 1.f, nullptr, 0);

    // --- softmax; T accumulation (rel-value collapsed); attn*V (WMMA) ---
    softmax_kernel<<<dim3(NHEAD * NPTS), dim3(256), 0, stream>>>(scores, wh);
    tsum_kernel<<<dim3(NPTS), dim3(256), 0, stream>>>(Urv, rv1_b, scores, T);
    for (int h = 0; h < NHEAD; ++h)
        gemm(wh + (size_t)h * NPTS * NPTS, NPTS, vt + (size_t)h * HDIM * NPTS, NPTS,
             ctx + h * HDIM, DMODEL, NPTS, HDIM, NPTS, 1.f, 0.f, nullptr, 0);
    ctx2_kernel<<<dim3(NPTS), dim3(DMODEL), 0, stream>>>(T, rv2_w, rv2_b, ctx);

    // --- residual, LN2, SwiGLU FFN (WMMA x3), final residual ---
    add_kernel<<<dim3(NPTS * DMODEL / 256), dim3(256), 0, stream>>>(x, ctx, x2, NPTS * DMODEL);
    layernorm_kernel<<<dim3(NPTS), dim3(DMODEL), 0, stream>>>(x2, ln2_w, ln2_b, xn2h);
    gemm(xn2h, 256, w1h, 256, h1, HIDP, NPTS, HIDP, 256, 1.f, 0.f, ffn_b1, HIDR);
    gemm(xn2h, 256, w2h, 256, h2, HIDP, NPTS, HIDP, 256, 1.f, 0.f, ffn_b2, HIDR);
    swiglu_kernel<<<dim3((NPTS * HIDP + 255) / 256), dim3(256), 0, stream>>>(h1, h2, hph, NPTS * HIDP);
    gemm(hph, HIDP, w3h, HIDP, ffnO, DMODEL, NPTS, DMODEL, HIDP, 1.f, 0.f, ffn_b3, DMODEL);
    add_kernel<<<dim3(NPTS * DMODEL / 256), dim3(256), 0, stream>>>(x2, ffnO, out, NPTS * DMODEL);
}
